// MoELayer_10840497455341
// MI455X (gfx1250) — matrix operand order repaired and checked by reading, not compiled, not run on hardware
//
#include <hip/hip_runtime.h>

typedef __attribute__((ext_vector_type(16))) __bf16 v16bf;
typedef __attribute__((ext_vector_type(8)))  float  v8f;

namespace {
constexpr int kB = 8192;   // batch tokens
constexpr int kD = 1024;   // d_model
constexpr int kH = 4096;   // expert hidden
constexpr int kE = 8;      // experts

constexpr int BM = 128, BN = 256, BK = 32;
constexpr int LDSU = BK / 2 + 4;   // uints per LDS row (16 data + 4 pad -> 80B stride)

static_assert(kB % BM == 0 && kH % BN == 0 && kD % BN == 0, "tile divisibility");
static_assert(kD % BK == 0 && kH % BK == 0, "K divisibility");
}

union Frag { unsigned u[8]; v16bf v; };

__device__ __forceinline__ unsigned pack2_bf16(float a, float b) {
  union { __bf16 h[2]; unsigned u; } p;
  p.h[0] = (__bf16)a;
  p.h[1] = (__bf16)b;
  return p.u;
}

__device__ __forceinline__ unsigned short to_bf16_bits(float a) {
  union { __bf16 h; unsigned short s; } p;
  p.h = (__bf16)a;
  return p.s;
}

// ---------------------------------------------------------------------------
// Gating: logits = x @ Wg^T ; softmax over E=8 ; keep top-3 (no renorm).
// One wave (32 lanes) per token row; Wg cached in LDS.
// ---------------------------------------------------------------------------
__global__ __launch_bounds__(256)
void moe_gating_kernel(const float* __restrict__ x,
                       const float* __restrict__ Wg,
                       float* __restrict__ gates) {
  __shared__ float sWg[kE * kD];  // 32 KB
  const int tid = threadIdx.x;
  for (int i = tid; i < kE * kD; i += 256) sWg[i] = Wg[i];
  __syncthreads();

  const int lane = tid & 31;
  const int wave = tid >> 5;
  const int row = blockIdx.x * 8 + wave;

  float acc[kE];
#pragma unroll
  for (int e = 0; e < kE; ++e) acc[e] = 0.f;

  const float* xr = x + (size_t)row * kD;
  for (int d = lane; d < kD; d += 32) {
    const float xv = xr[d];
#pragma unroll
    for (int e = 0; e < kE; ++e) acc[e] += xv * sWg[e * kD + d];
  }
#pragma unroll
  for (int e = 0; e < kE; ++e) {
#pragma unroll
    for (int off = 16; off > 0; off >>= 1)
      acc[e] += __shfl_xor(acc[e], off, 32);
  }

  if (lane == 0) {
    float mx = acc[0];
#pragma unroll
    for (int e = 1; e < kE; ++e) mx = fmaxf(mx, acc[e]);
    float g[kE];
    float s = 0.f;
#pragma unroll
    for (int e = 0; e < kE; ++e) { g[e] = __expf(acc[e] - mx); s += g[e]; }
    const float inv = 1.f / s;
#pragma unroll
    for (int e = 0; e < kE; ++e) g[e] *= inv;
    // top-3 selection threshold (softmax outputs are > 0)
    float v1 = -1.f, v2 = -1.f, v3 = -1.f;
#pragma unroll
    for (int e = 0; e < kE; ++e) {
      const float v = g[e];
      if (v > v1)      { v3 = v2; v2 = v1; v1 = v; }
      else if (v > v2) { v3 = v2; v2 = v; }
      else if (v > v3) { v3 = v; }
    }
#pragma unroll
    for (int e = 0; e < kE; ++e)
      gates[(size_t)row * kE + e] = (g[e] >= v3) ? g[e] : 0.f;
  }
}

// ---------------------------------------------------------------------------
// Tiled bf16-WMMA GEMM, C[M,N] = A[M,K] * B[N,K]^T (both K-major in memory).
// MODE 0: A = x (fp32), epilogue relu(acc + b1) -> h (bf16)
// MODE 1: A = h (bf16), epilogue gate * (acc + b2) (+=) -> out (fp32)
// Block tile 128x256x32, 256 threads = 8 waves (2x4), each wave 64x64:
// per K-step a wave does 16 ds_load_b128 feeding 16 v_wmma (1:1 ratio).
// KDIM/LDOUT are compile-time so all strides are shifts.
// ---------------------------------------------------------------------------
template <int MODE, int KDIM, int LDOUT>
__global__ __launch_bounds__(256)
void moe_gemm_kernel(const float* __restrict__ A32g,
                     const unsigned short* __restrict__ A16g,
                     const float* __restrict__ Bwg,
                     const float* __restrict__ bias,
                     const float* __restrict__ gates, int expert,
                     float* __restrict__ outF,
                     unsigned short* __restrict__ outH,
                     int accumulate) {
  __shared__ unsigned ldsA[2][BM * LDSU];  // 2 x 10 KB
  __shared__ unsigned ldsB[2][BN * LDSU];  // 2 x 20 KB

  const int tid = threadIdx.x;
  const int blockN = blockIdx.x * BN;
  const int blockM = blockIdx.y * BM;

  const float* Bsrc = Bwg + (size_t)blockN * KDIM;
  const float* A32 = nullptr;
  const unsigned short* A16 = nullptr;
  if constexpr (MODE == 0) A32 = A32g + (size_t)blockM * KDIM;
  else                     A16 = A16g + (size_t)blockM * KDIM;

  const int lane = tid & 31;
  const int wave = tid >> 5;
  const int wm = wave & 1;   // 2 wave-rows of 64
  const int wn = wave >> 1;  // 4 wave-cols of 64
  const int mB = wm * 64;
  const int nB = wn * 64;
  const int lm = lane & 15;
  const int hi = lane >> 4;

  v8f acc[4][4];
#pragma unroll
  for (int mi = 0; mi < 4; ++mi)
#pragma unroll
    for (int ni = 0; ni < 4; ++ni)
#pragma unroll
      for (int q = 0; q < 8; ++q) acc[mi][ni][q] = 0.f;

  constexpr int KT = KDIM / BK;

  float4 aReg[4];
  uint4  aRegH[2];
  float4 bReg[8];

  auto loadA = [&](int k0) {
    if constexpr (MODE == 0) {
#pragma unroll
      for (int i = 0; i < 4; ++i) {
        const int idx = tid + i * 256;
        const int row = idx >> 3, seg = idx & 7;
        aReg[i] = *(const float4*)(A32 + (size_t)row * KDIM + k0 + seg * 4);
      }
    } else {
#pragma unroll
      for (int i = 0; i < 2; ++i) {
        const int idx = tid + i * 256;
        const int row = idx >> 2, seg = idx & 3;
        aRegH[i] = *(const uint4*)(A16 + (size_t)row * KDIM + k0 + seg * 8);
      }
    }
  };
  auto loadB = [&](int k0) {
#pragma unroll
    for (int i = 0; i < 8; ++i) {
      const int idx = tid + i * 256;
      const int row = idx >> 3, seg = idx & 7;
      bReg[i] = *(const float4*)(Bsrc + (size_t)row * KDIM + k0 + seg * 4);
    }
  };
  auto stash = [&](int buf) {
    unsigned* As = ldsA[buf];
    unsigned* Bs = ldsB[buf];
    if constexpr (MODE == 0) {
#pragma unroll
      for (int i = 0; i < 4; ++i) {
        const int idx = tid + i * 256;
        const int row = idx >> 3, seg = idx & 7;
        As[row * LDSU + seg * 2 + 0] = pack2_bf16(aReg[i].x, aReg[i].y);
        As[row * LDSU + seg * 2 + 1] = pack2_bf16(aReg[i].z, aReg[i].w);
      }
    } else {
#pragma unroll
      for (int i = 0; i < 2; ++i) {
        const int idx = tid + i * 256;
        const int row = idx >> 2, seg = idx & 3;
        As[row * LDSU + seg * 4 + 0] = aRegH[i].x;
        As[row * LDSU + seg * 4 + 1] = aRegH[i].y;
        As[row * LDSU + seg * 4 + 2] = aRegH[i].z;
        As[row * LDSU + seg * 4 + 3] = aRegH[i].w;
      }
    }
#pragma unroll
    for (int i = 0; i < 8; ++i) {
      const int idx = tid + i * 256;
      const int row = idx >> 3, seg = idx & 7;
      Bs[row * LDSU + seg * 2 + 0] = pack2_bf16(bReg[i].x, bReg[i].y);
      Bs[row * LDSU + seg * 2 + 1] = pack2_bf16(bReg[i].z, bReg[i].w);
    }
  };

  loadA(0);
  loadB(0);
  stash(0);
  __syncthreads();

  for (int kt = 0; kt < KT; ++kt) {
    const int cur = kt & 1;
    const bool hasNext = (kt + 1 < KT);
    if (hasNext) { loadA((kt + 1) * BK); loadB((kt + 1) * BK); }
    if (kt + 2 < KT) {  // emits global_prefetch_b8 for the tile after next
      const size_t kp = (size_t)(kt + 2) * BK;
      __builtin_prefetch(Bsrc + (size_t)tid * KDIM + kp, 0, 1);
      const int arow = tid >> 1;
      if constexpr (MODE == 0) __builtin_prefetch(A32 + (size_t)arow * KDIM + kp, 0, 1);
      else                     __builtin_prefetch(A16 + (size_t)arow * KDIM + kp, 0, 1);
    }

    const unsigned* As = ldsA[cur];
    const unsigned* Bs = ldsB[cur];
    Frag af[4], bf[4];
    // A 16x32 bf16 layout (ISA 7.12.2): lane=M(0..15)+half, uint idx v+(v&4)+4*hi
#pragma unroll
    for (int mi = 0; mi < 4; ++mi)
#pragma unroll
      for (int v = 0; v < 8; ++v)
        af[mi].u[v] = As[(mB + mi * 16 + lm) * LDSU + v + (v & 4) + 4 * hi];
    // B 32x16 bf16 layout: lane=N(0..15)+half, uint idx v+(v&4)+4*hi (same K order as A)
#pragma unroll
    for (int ni = 0; ni < 4; ++ni)
#pragma unroll
      for (int v = 0; v < 8; ++v)
        bf[ni].u[v] = Bs[(nB + ni * 16 + lm) * LDSU + v + (v & 4) + 4 * hi];

#pragma unroll
    for (int mi = 0; mi < 4; ++mi)
#pragma unroll
      for (int ni = 0; ni < 4; ++ni)
        acc[mi][ni] = __builtin_amdgcn_wmma_f32_16x16x32_bf16(
            false, af[mi].v, false, bf[ni].v, (short)0, acc[mi][ni], false, false);

    if (hasNext) stash(cur ^ 1);
    __syncthreads();
  }

  // Epilogue. C layout: lanes 0-15 N=lane, M=r; lanes 16-31 N=lane-16, M=r+8.
#pragma unroll
  for (int ni = 0; ni < 4; ++ni) {
    const int gc = blockN + nB + ni * 16 + lm;
    const float bv = bias[gc];
#pragma unroll
    for (int mi = 0; mi < 4; ++mi) {
#pragma unroll
      for (int r = 0; r < 8; ++r) {
        const int gr = blockM + mB + mi * 16 + hi * 8 + r;
        const float v = acc[mi][ni][r] + bv;
        if constexpr (MODE == 0) {
          outH[(size_t)gr * LDOUT + gc] = to_bf16_bits(fmaxf(v, 0.f));
        } else {
          const float g = gates[(size_t)gr * kE + expert];
          const float gv = g * v;
          float* o = outF + (size_t)gr * LDOUT + gc;
          if (accumulate) *o += gv;
          else            *o = gv;
        }
      }
    }
  }
}

// ---------------------------------------------------------------------------
// Launch: gating, then per-expert (GEMM1 -> h bf16 workspace, GEMM2 -> out).
// Workspace: [gates: B*E f32 = 256KB][h: B*H bf16 = 64MB] (~64.3 MB used).
// h is L2-resident (192 MB L2), so the inter-GEMM round-trip is nearly free.
// ---------------------------------------------------------------------------
extern "C" void kernel_launch(void* const* d_in, const int* in_sizes, int n_in,
                              void* d_out, int out_size, void* d_ws, size_t ws_size,
                              hipStream_t stream) {
  (void)in_sizes; (void)n_in; (void)out_size; (void)ws_size;
  const float* x  = (const float*)d_in[0];
  const float* Wg = (const float*)d_in[1];
  const float* W1 = (const float*)d_in[2];
  const float* b1 = (const float*)d_in[3];
  const float* W2 = (const float*)d_in[4];
  const float* b2 = (const float*)d_in[5];
  float* out = (float*)d_out;

  float* gates = (float*)d_ws;
  unsigned short* hbuf =
      (unsigned short*)((char*)d_ws + (size_t)kB * kE * sizeof(float));

  moe_gating_kernel<<<kB / 8, 256, 0, stream>>>(x, Wg, gates);

  for (int e = 0; e < kE; ++e) {
    // h = relu(x @ W1[e]^T + b1[e])   [B, H] bf16
    moe_gemm_kernel<0, kD, kH><<<dim3(kH / BN, kB / BM), 256, 0, stream>>>(
        x, nullptr,
        W1 + (size_t)e * kH * kD, b1 + (size_t)e * kH,
        nullptr, e, nullptr, hbuf,
        0);
    // out (+)= gate[:,e] * (h @ W2[e]^T + b2[e])   [B, D] f32
    moe_gemm_kernel<1, kH, kD><<<dim3(kD / BN, kB / BM), 256, 0, stream>>>(
        nullptr, hbuf,
        W2 + (size_t)e * kD * kH, b2 + (size_t)e * kD,
        gates, e, out, nullptr,
        e != 0);
  }
}
